// CustomBigBirdModel_3246995276325
// MI455X (gfx1250) — matrix operand order repaired and checked by reading, not compiled, not run on hardware
//
#include <hip/hip_runtime.h>

// ---------------- model dims (compile-time) ----------------
constexpr int kB   = 2;
constexpr int kS   = 4096;
constexpr int kIN  = 1280;
constexpr int kD   = 768;
constexpr int kFF  = 3072;
constexpr int kH   = 12;
constexpr int kDH  = 64;
constexpr int kBS  = 64;            // attention block size
constexpr int kNB  = kS / kBS;      // 64 key blocks
constexpr int kNA  = 8;             // gathered blocks per interior q block
constexpr int kNI  = kNB - 2;       // 62 interior q blocks
constexpr int kL   = 12;
constexpr int kM   = kB * kS;       // 8192 token rows
constexpr int kC   = 2;

typedef __bf16 bf16;
typedef __attribute__((ext_vector_type(16))) __bf16 v16bf;
typedef __attribute__((ext_vector_type(8)))  float  v8f;
typedef __attribute__((ext_vector_type(4)))  unsigned int v4u;
typedef __attribute__((ext_vector_type(8)))  int v8i;
typedef __attribute__((ext_vector_type(4)))  int v4i;

union FragBF { v16bf v; uint4 q[2]; };
union BfBits { unsigned short s; bf16 b; };
union U4Bits { uint4 q; unsigned short u[8]; };

__device__ inline v8f zero8() {
  v8f z;
#pragma unroll
  for (int i = 0; i < 8; ++i) z[i] = 0.f;
  return z;
}

// A-operand fragment (ISA 7.12.2, 16-bit A 16x32): lane half h, elements
// 0..7 -> K = h*8.. ; elements 8..15 -> K = 16+h*8.. ; p = &row[h*8].
__device__ inline v16bf ld_frag_a(const bf16* p) {
  FragBF f;
  f.q[0] = *(const uint4*)(p);
  f.q[1] = *(const uint4*)(p + 16);
  return f.v;
}
// B-operand fragment: lane holds n fixed, same K order as A; p = &row[h*8].
__device__ inline v16bf ld_frag_b(const bf16* p) {
  FragBF f;
  f.q[0] = *(const uint4*)(p);
  f.q[1] = *(const uint4*)(p + 16);
  return f.v;
}

__device__ inline v8f wmma_bf16(v16bf a, v16bf b, v8f c) {
  return __builtin_amdgcn_wmma_f32_16x16x32_bf16(false, a, false, b,
                                                 (short)0, c, false, false);
}

__device__ inline float gelu_exact(float x) {
  return 0.5f * x * (1.0f + erff(x * 0.7071067811865475f));
}

// =====================================================================
// Tensor Data Mover: 2D tile load Global(bf16, contiguous rows) -> LDS,
// LDS rows padded via D# pad fields.  (CDNA5 ISA)
//   pad_interval code i => pad after 2^(i+1) DWORDs ; pad_amount code a =>
//   a+1 DWORDs inserted.
// =====================================================================
#define HAVE_TDM __has_builtin(__builtin_amdgcn_tensor_load_to_lds)

#if HAVE_TDM
__device__ inline void tdm_load_tile_2d(unsigned lds_addr, const void* gptr,
                                        unsigned row_elems, unsigned rows,
                                        unsigned stride_elems,
                                        unsigned padi, unsigned pada)
{
  const unsigned long long ga = (unsigned long long)(size_t)gptr;
  v4u g0;
  g0[0] = 1u;                                            // count=1, user mode
  g0[1] = lds_addr;                                      // LDS byte address
  g0[2] = (unsigned)(ga & 0xFFFFFFFFu);                  // global addr lo
  g0[3] = (unsigned)((ga >> 32) & 0x01FFFFFFu) | (2u << 30);  // hi | type=2
  v8i g1;
  g1[0] = (int)((1u << 16) |            // data_size = 1 (2 bytes)
                (1u << 20) |            // pad_enable
                (padi << 22) |          // pad_interval
                (pada << 25));          // pad_amount
  g1[1] = (int)((row_elems & 0xFFFFu) << 16);                     // dim0 lo
  g1[2] = (int)((row_elems >> 16) & 0xFFFFu) |
          (int)((rows & 0xFFFFu) << 16);                          // dim0 hi | dim1 lo
  g1[3] = (int)((rows >> 16) & 0xFFFFu) |
          (int)((row_elems & 0xFFFFu) << 16);                     // dim1 hi | tile_dim0
  g1[4] = (int)(rows & 0xFFFFu);                                  // tile_dim1 (tile_dim2=0)
  g1[5] = (int)stride_elems;                                      // dim0_stride lo32
  g1[6] = 0;
  g1[7] = 0;
  v4i z4 = {0, 0, 0, 0};
#if defined(__clang_major__) && __clang_major__ >= 23
  v8i z8 = {0, 0, 0, 0, 0, 0, 0, 0};
  __builtin_amdgcn_tensor_load_to_lds(g0, g1, z4, z4, z8, 0);
#else
  __builtin_amdgcn_tensor_load_to_lds(g0, g1, z4, z4, 0);
#endif
}
#endif

// =====================================================================
// GEMM: C[M,N] = A[M,K](bf16) @ BT[N,K](bf16, pre-transposed) + bias(f32)
// Tile 128x128, 8 waves (2x4), each wave 64x32 (4x2 accs), BK=32.
// EPI: 0 = bias -> f32 ; 1 = bias -> bf16 ; 2 = bias+GELU -> bf16
// =====================================================================
template<int EPI>
__global__ void __launch_bounds__(256)
gemm_bf16_wmma(const bf16* __restrict__ A, const bf16* __restrict__ BT,
               const float* __restrict__ bias, void* __restrict__ Cout,
               int M, int N, int K)
{
  __shared__ bf16 As[128][40];   // [m][k], 80B row stride (16 DW + 4 DW pad)
  __shared__ bf16 Bs[128][40];   // [n][k]

  const int tid  = threadIdx.x;
  const int lane = tid & 31;
  const int w    = tid >> 5;
  const int wm   = w & 1;
  const int wn   = w >> 1;
  const int half = lane >> 4;
  const int l15  = lane & 15;
  const int bm   = blockIdx.y * 128;
  const int bn   = blockIdx.x * 128;

#if HAVE_TDM
  const unsigned ldsA = (unsigned)(size_t)(void*)&As[0][0];
  const unsigned ldsB = (unsigned)(size_t)(void*)&Bs[0][0];
#endif

  v8f acc[4][2];
#pragma unroll
  for (int i = 0; i < 4; ++i) { acc[i][0] = zero8(); acc[i][1] = zero8(); }

  for (int k0 = 0; k0 < K; k0 += 32) {
#if HAVE_TDM
    if (w == 0) {
      tdm_load_tile_2d(ldsA, A  + (size_t)bm * K + k0, 32, 128, (unsigned)K, 3, 3);
      tdm_load_tile_2d(ldsB, BT + (size_t)bn * K + k0, 32, 128, (unsigned)K, 3, 3);
      __builtin_amdgcn_s_wait_tensorcnt((short)0);
    }
#else
    // manual staging: 128 rows x 32 bf16 = 512 uint4 per tile, 2/thread
#pragma unroll
    for (int i = 0; i < 2; ++i) {
      const int f  = i * 256 + tid;
      const int r  = f >> 2;
      const int c8 = (f & 3) << 3;
      *(uint4*)&As[r][c8] = *(const uint4*)(A  + (size_t)(bm + r) * K + k0 + c8);
      *(uint4*)&Bs[r][c8] = *(const uint4*)(BT + (size_t)(bn + r) * K + k0 + c8);
    }
#endif
    __syncthreads();

    v16bf bfr[2];
#pragma unroll
    for (int fn = 0; fn < 2; ++fn)
      bfr[fn] = ld_frag_b(&Bs[wn * 32 + fn * 16 + l15][half * 8]);
#pragma unroll
    for (int fm = 0; fm < 4; ++fm) {
      v16bf af = ld_frag_a(&As[wm * 64 + fm * 16 + l15][half * 8]);
      acc[fm][0] = wmma_bf16(af, bfr[0], acc[fm][0]);
      acc[fm][1] = wmma_bf16(af, bfr[1], acc[fm][1]);
    }
    __syncthreads();
  }

  // epilogue: C-layout row = half*8 + elem, col = l15
#pragma unroll
  for (int fm = 0; fm < 4; ++fm) {
#pragma unroll
    for (int fn = 0; fn < 2; ++fn) {
      const int col  = bn + wn * 32 + fn * 16 + l15;
      const int row0 = bm + wm * 64 + fm * 16 + half * 8;
      const float bv = bias[col];
#pragma unroll
      for (int j = 0; j < 8; ++j) {
        float v = acc[fm][fn][j] + bv;
        if (EPI == 2) v = gelu_exact(v);
        if (EPI == 0) ((float*)Cout)[(size_t)(row0 + j) * N + col] = v;
        else          ((bf16*)Cout)[(size_t)(row0 + j) * N + col] = (bf16)v;
      }
    }
  }
}

// =====================================================================
// BigBird attention (bf16 q/k/v in, bf16 ctx out). One workgroup per
// (b, h, q-block); 8 waves 4x2 over the 64x64 ctx tile; streaming softmax.
// =====================================================================
template<bool EDGE>
__global__ void __launch_bounds__(256)
bigbird_attn(const bf16* __restrict__ Q, const bf16* __restrict__ Kc,
             const bf16* __restrict__ Vc, bf16* __restrict__ Ctx,
             const int* __restrict__ amask,
             const int* __restrict__ blk_idx,
             const unsigned char* __restrict__ blk_valid)
{
  __shared__ bf16 Qs[64][72];        // [m][d]   (32 DW + 4 DW pad per row)
  __shared__ bf16 Ks[64][72];        // [s'][d]
  __shared__ bf16 Vt[64][72];        // [d][s']  (transposed)
  __shared__ bf16 Ps[8][16][72];     // per-wave P tile [row][key]
  __shared__ float mb[64];

  int id = blockIdx.x;
  const int nq = EDGE ? 2 : kNI;
  const int qi = id % nq; id /= nq;
  const int h  = id % kH;
  const int b  = id / kH;
  const int qblk = EDGE ? (qi == 0 ? 0 : kNB - 1) : (qi + 1);

  const int tid  = threadIdx.x;
  const int lane = tid & 31;
  const int w    = tid >> 5;
  const int wm   = w >> 1;
  const int wn   = w & 1;
  const int half = lane >> 4;
  const int l15  = lane & 15;
  const size_t baseRow = (size_t)b * kS;
  const int hc = h * kDH;

  // ---- stage Q block (64x64 bf16) ----
#if HAVE_TDM
  if (w == 0)
    tdm_load_tile_2d((unsigned)(size_t)(void*)&Qs[0][0],
                     Q + (baseRow + qblk * kBS) * kD + hc, 64, 64, kD, 4, 3);
#else
#pragma unroll
  for (int i = 0; i < 2; ++i) {
    const int f = i * 256 + tid;
    const int r = f >> 3, c8 = (f & 7) << 3;
    *(uint4*)&Qs[r][c8] = *(const uint4*)(Q + (baseRow + qblk * kBS + r) * kD + hc + c8);
  }
#endif

  float mrun[8], lrun[8];
#pragma unroll
  for (int e = 0; e < 8; ++e) { mrun[e] = -1e30f; lrun[e] = 0.f; }
  v8f cacc[2] = { zero8(), zero8() };

  const float scale = 0.125f;
  const int nkb = EDGE ? kNB : kNA;

  for (int j = 0; j < nkb; ++j) {
    int kb; bool valid;
    if (EDGE) { kb = j; valid = true; }
    else      { kb = blk_idx[qi * kNA + j]; valid = blk_valid[qi * kNA + j] != 0; }

    __syncthreads();   // previous iteration done with Ks/Vt/mb
#if HAVE_TDM
    if (w == 0)
      tdm_load_tile_2d((unsigned)(size_t)(void*)&Ks[0][0],
                       Kc + (baseRow + kb * kBS) * kD + hc, 64, 64, kD, 4, 3);
#else
#pragma unroll
    for (int i = 0; i < 2; ++i) {
      const int f = i * 256 + tid;
      const int r = f >> 3, c8 = (f & 7) << 3;
      *(uint4*)&Ks[r][c8] = *(const uint4*)(Kc + (baseRow + kb * kBS + r) * kD + hc + c8);
    }
#endif
    // V staged transposed (TDM cannot transpose): 8 bf16 per uint4, scatter
#pragma unroll
    for (int i = 0; i < 2; ++i) {
      const int f = i * 256 + tid;
      const int r = f >> 3, c8 = (f & 7) << 3;
      U4Bits t; t.q = *(const uint4*)(Vc + (baseRow + kb * kBS + r) * kD + hc + c8);
#pragma unroll
      for (int jj = 0; jj < 8; ++jj) {
        BfBits bb; bb.s = t.u[jj];
        Vt[c8 + jj][r] = bb.b;
      }
    }
    if (tid < 64) {
      const int s = kb * kBS + tid;
      const bool ok = valid && (amask[b * kS + s] > 0);
      mb[tid] = ok ? 0.f : -1e9f;
    }
#if HAVE_TDM
    if (w == 0) __builtin_amdgcn_s_wait_tensorcnt((short)0);
#endif
    __syncthreads();

    // S = Q @ K^T : wave's 16 rows x 64 keys -> 4 accumulators
    v8f sacc[4] = { zero8(), zero8(), zero8(), zero8() };
#pragma unroll
    for (int kk = 0; kk < 2; ++kk) {
      v16bf aq = ld_frag_a(&Qs[wm * 16 + l15][kk * 32 + half * 8]);
#pragma unroll
      for (int c = 0; c < 4; ++c) {
        v16bf bk = ld_frag_b(&Ks[c * 16 + l15][kk * 32 + half * 8]);
        sacc[c] = wmma_bf16(aq, bk, sacc[c]);
      }
    }

    // streaming softmax (row = half*8 + e ; col = c*16 + l15)
#pragma unroll
    for (int e = 0; e < 8; ++e) {
      float sv[4];
      float rmax = -1e30f;
#pragma unroll
      for (int c = 0; c < 4; ++c) {
        sv[c] = sacc[c][e] * scale + mb[c * 16 + l15];
        rmax = fmaxf(rmax, sv[c]);
      }
#pragma unroll
      for (int m = 1; m <= 8; m <<= 1)
        rmax = fmaxf(rmax, __shfl_xor(rmax, m, 32));
      const float newm = fmaxf(mrun[e], rmax);
      const float corr = expf(mrun[e] - newm);
      float psum = 0.f;
#pragma unroll
      for (int c = 0; c < 4; ++c) {
        float p = expf(sv[c] - newm);
        sacc[c][e] = p;
        psum += p;
      }
#pragma unroll
      for (int m = 1; m <= 8; m <<= 1)
        psum += __shfl_xor(psum, m, 32);
      lrun[e] = lrun[e] * corr + psum;
      mrun[e] = newm;
      cacc[0][e] *= corr;
      cacc[1][e] *= corr;
    }

    // write P (C-layout) into per-wave LDS tile in A-layout [row][key]
#pragma unroll
    for (int c = 0; c < 4; ++c)
#pragma unroll
      for (int e = 0; e < 8; ++e)
        Ps[w][half * 8 + e][c * 16 + l15] = (bf16)sacc[c][e];
    asm volatile("s_wait_dscnt 0" ::: "memory");   // intra-wave LDS RAW

    // ctx += P @ V
#pragma unroll
    for (int kk = 0; kk < 2; ++kk) {
      v16bf ap = ld_frag_a(&Ps[w][l15][kk * 32 + half * 8]);
#pragma unroll
      for (int fn = 0; fn < 2; ++fn) {
        v16bf bv = ld_frag_b(&Vt[wn * 32 + fn * 16 + l15][kk * 32 + half * 8]);
        cacc[fn] = wmma_bf16(ap, bv, cacc[fn]);
      }
    }
  }

  // normalize and store ctx (bf16, col = h*64 + d)
#pragma unroll
  for (int fn = 0; fn < 2; ++fn)
#pragma unroll
    for (int e = 0; e < 8; ++e) {
      const float inv = 1.f / lrun[e];
      const int row = qblk * kBS + wm * 16 + half * 8 + e;
      const int col = hc + wn * 32 + fn * 16 + l15;
      Ctx[(baseRow + row) * (size_t)kD + col] = (bf16)(cacc[fn][e] * inv);
    }
}

// =====================================================================
// LayerNorm over D=768; writes fp32 result + bf16 mirror for next GEMM.
// =====================================================================
__global__ void __launch_bounds__(256)
layernorm_kernel(const float* __restrict__ base, const float* __restrict__ resid,
                 const float* __restrict__ pos, const float* __restrict__ tok,
                 const float* __restrict__ g, const float* __restrict__ be,
                 float* __restrict__ out, bf16* __restrict__ out_bf)
{
  __shared__ float red[256];
  const int row = blockIdx.x;
  const int s   = row & (kS - 1);
  const int tid = threadIdx.x;
  const size_t b0 = (size_t)row * kD;

  float v[3];
  float sum = 0.f;
#pragma unroll
  for (int i = 0; i < 3; ++i) {
    const int c = tid + i * 256;
    float x = base[b0 + c];
    if (resid) x += resid[b0 + c];
    if (pos)   x += pos[(size_t)s * kD + c];
    if (tok)   x += tok[c];
    v[i] = x; sum += x;
  }
  red[tid] = sum; __syncthreads();
  for (int off = 128; off > 0; off >>= 1) {
    if (tid < off) red[tid] += red[tid + off];
    __syncthreads();
  }
  const float mean = red[0] * (1.f / kD);
  __syncthreads();

  float vs = 0.f;
#pragma unroll
  for (int i = 0; i < 3; ++i) { const float d = v[i] - mean; vs += d * d; }
  red[tid] = vs; __syncthreads();
  for (int off = 128; off > 0; off >>= 1) {
    if (tid < off) red[tid] += red[tid + off];
    __syncthreads();
  }
  const float rstd = rsqrtf(red[0] * (1.f / kD) + 1e-12f);

#pragma unroll
  for (int i = 0; i < 3; ++i) {
    const int c = tid + i * 256;
    const float y = (v[i] - mean) * rstd * g[c] + be[c];
    out[b0 + c] = y;
    out_bf[b0 + c] = (bf16)y;
  }
}

// =====================================================================
// fp32 -> bf16 flat conversion (for the raw embedding input)
// =====================================================================
__global__ void __launch_bounds__(256)
convert_kernel(const float* __restrict__ in, bf16* __restrict__ out, long long n)
{
  const long long i = ((long long)blockIdx.x * 256 + threadIdx.x) * 4;
  if (i + 3 < n) {
    const float4 t = *(const float4*)(in + i);
    out[i + 0] = (bf16)t.x; out[i + 1] = (bf16)t.y;
    out[i + 2] = (bf16)t.z; out[i + 3] = (bf16)t.w;
  }
}

// =====================================================================
// Weight convert + transpose: fp32 [K][N] -> bf16 [N][K] (GEMM B layout)
// =====================================================================
__global__ void __launch_bounds__(256)
wtrans_kernel(const float* __restrict__ in, bf16* __restrict__ out, int K, int N)
{
  __shared__ float t[64][65];
  const int nb = blockIdx.x * 64, kb = blockIdx.y * 64;
  const int tid = threadIdx.x;
#pragma unroll
  for (int i = 0; i < 16; ++i) {
    const int f = i * 256 + tid;
    const int r = f >> 6, c = f & 63;
    t[r][c] = in[(size_t)(kb + r) * N + nb + c];
  }
  __syncthreads();
#pragma unroll
  for (int i = 0; i < 16; ++i) {
    const int f = i * 256 + tid;
    const int r = f >> 6, c = f & 63;
    out[(size_t)(nb + r) * K + kb + c] = (bf16)t[c][r];
  }
}

// =====================================================================
// Classifier: out[b,:] = x[b*S, :] @ Wc + bc
// =====================================================================
__global__ void __launch_bounds__(256)
classifier_kernel(const float* __restrict__ x, const float* __restrict__ Wc,
                  const float* __restrict__ bc, float* __restrict__ out)
{
  __shared__ float r0[256], r1[256];
  const int b = blockIdx.x;
  const int tid = threadIdx.x;
  const float* xr = x + (size_t)b * kS * kD;
  float a0 = 0.f, a1 = 0.f;
  for (int c = tid; c < kD; c += 256) {
    const float xv = xr[c];
    a0 += xv * Wc[c * kC + 0];
    a1 += xv * Wc[c * kC + 1];
  }
  r0[tid] = a0; r1[tid] = a1; __syncthreads();
  for (int off = 128; off > 0; off >>= 1) {
    if (tid < off) { r0[tid] += r0[tid + off]; r1[tid] += r1[tid + off]; }
    __syncthreads();
  }
  if (tid == 0) {
    out[b * kC + 0] = r0[0] + bc[0];
    out[b * kC + 1] = r1[0] + bc[1];
  }
}

// =====================================================================
extern "C" void kernel_launch(void* const* d_in, const int* in_sizes, int n_in,
                              void* d_out, int out_size, void* d_ws, size_t ws_size,
                              hipStream_t stream)
{
  (void)in_sizes; (void)n_in; (void)out_size; (void)ws_size;
  const float* emb   = (const float*)d_in[0];
  const int*   amask = (const int*)  d_in[1];
  const float* Wp    = (const float*)d_in[2];
  const float* bp    = (const float*)d_in[3];
  const float* pos   = (const float*)d_in[4];
  const float* tok   = (const float*)d_in[5];
  const float* lneg  = (const float*)d_in[6];
  const float* lneb  = (const float*)d_in[7];
  const float* Wq    = (const float*)d_in[8];
  const float* bq    = (const float*)d_in[9];
  const float* Wk    = (const float*)d_in[10];
  const float* bk    = (const float*)d_in[11];
  const float* Wv    = (const float*)d_in[12];
  const float* bv    = (const float*)d_in[13];
  const float* Wo    = (const float*)d_in[14];
  const float* bo    = (const float*)d_in[15];
  const float* l1g   = (const float*)d_in[16];
  const float* l1b   = (const float*)d_in[17];
  const float* W1    = (const float*)d_in[18];
  const float* b1    = (const float*)d_in[19];
  const float* W2    = (const float*)d_in[20];
  const float* b2    = (const float*)d_in[21];
  const float* l2g   = (const float*)d_in[22];
  const float* l2b   = (const float*)d_in[23];
  const float* Wc    = (const float*)d_in[24];
  const float* bc    = (const float*)d_in[25];
  const int*   bidx  = (const int*)  d_in[26];
  const unsigned char* bval = (const unsigned char*)d_in[27];

  // ---- workspace carving (256B aligned chunks) ----
  char* wsb = (char*)d_ws;
  size_t off = 0;
  auto carve = [&](size_t bytes) {
    void* p = wsb + off;
    off += (bytes + 255) & ~(size_t)255;
    return p;
  };
  const size_t MD = (size_t)kM * kD;
  const size_t MF = (size_t)kM * kFF;
  float* xA  = (float*)carve(MD * 4);
  float* xB  = (float*)carve(MD * 4);
  float* t0  = (float*)carve(MD * 4);
  bf16* ebf  = (bf16*)carve((size_t)kM * kIN * 2);
  bf16* xbf  = (bf16*)carve(MD * 2);
  bf16* qb   = (bf16*)carve(MD * 2);
  bf16* kbf  = (bf16*)carve(MD * 2);
  bf16* vbf  = (bf16*)carve(MD * 2);
  bf16* ctx  = (bf16*)carve(MD * 2);
  bf16* h1   = (bf16*)carve(MF * 2);
  bf16* WpT  = (bf16*)carve((size_t)kIN * kD * 2);
  const size_t Wdd = (size_t)kD * kD;
  const size_t Wdf = (size_t)kD * kFF;
  bf16* WqT  = (bf16*)carve(kL * Wdd * 2);
  bf16* WkT  = (bf16*)carve(kL * Wdd * 2);
  bf16* WvT  = (bf16*)carve(kL * Wdd * 2);
  bf16* WoT  = (bf16*)carve(kL * Wdd * 2);
  bf16* W1T  = (bf16*)carve(kL * Wdf * 2);
  bf16* W2T  = (bf16*)carve(kL * Wdf * 2);

  const dim3 blk(256);
  const dim3 gD(kD / 128, kM / 128);
  const dim3 gFF(kFF / 128, kM / 128);

  // ---- one-time-per-call weight/activation conversion ----
  convert_kernel<<<dim3((unsigned)((size_t)kM * kIN / 1024)), blk, 0, stream>>>(
      emb, ebf, (long long)kM * kIN);
  wtrans_kernel<<<dim3(kD / 64, kIN / 64), blk, 0, stream>>>(Wp, WpT, kIN, kD);
  for (int l = 0; l < kL; ++l) {
    wtrans_kernel<<<dim3(kD / 64, kD / 64), blk, 0, stream>>>(Wq + l * Wdd, WqT + l * Wdd, kD, kD);
    wtrans_kernel<<<dim3(kD / 64, kD / 64), blk, 0, stream>>>(Wk + l * Wdd, WkT + l * Wdd, kD, kD);
    wtrans_kernel<<<dim3(kD / 64, kD / 64), blk, 0, stream>>>(Wv + l * Wdd, WvT + l * Wdd, kD, kD);
    wtrans_kernel<<<dim3(kD / 64, kD / 64), blk, 0, stream>>>(Wo + l * Wdd, WoT + l * Wdd, kD, kD);
    wtrans_kernel<<<dim3(kFF / 64, kD / 64), blk, 0, stream>>>(W1 + l * Wdf, W1T + l * Wdf, kD, kFF);
    wtrans_kernel<<<dim3(kD / 64, kFF / 64), blk, 0, stream>>>(W2 + l * Wdf, W2T + l * Wdf, kFF, kD);
  }

  // ---- embedding projection + embed-LN ----
  gemm_bf16_wmma<0><<<gD, blk, 0, stream>>>(ebf, WpT, bp, t0, kM, kD, kIN);
  layernorm_kernel<<<kM, blk, 0, stream>>>(t0, nullptr, pos, tok, lneg, lneb, xA, xbf);

  float* xc = xA;
  float* xn = xB;
  for (int l = 0; l < kL; ++l) {
    gemm_bf16_wmma<1><<<gD, blk, 0, stream>>>(xbf, WqT + l * Wdd, bq + l * kD, qb,  kM, kD, kD);
    gemm_bf16_wmma<1><<<gD, blk, 0, stream>>>(xbf, WkT + l * Wdd, bk + l * kD, kbf, kM, kD, kD);
    gemm_bf16_wmma<1><<<gD, blk, 0, stream>>>(xbf, WvT + l * Wdd, bv + l * kD, vbf, kM, kD, kD);

    bigbird_attn<false><<<dim3(kB * kH * kNI), blk, 0, stream>>>(qb, kbf, vbf, ctx, amask, bidx, bval);
    bigbird_attn<true ><<<dim3(kB * kH * 2),   blk, 0, stream>>>(qb, kbf, vbf, ctx, amask, bidx, bval);

    gemm_bf16_wmma<0><<<gD, blk, 0, stream>>>(ctx, WoT + l * Wdd, bo + l * kD, t0, kM, kD, kD);
    layernorm_kernel<<<kM, blk, 0, stream>>>(t0, xc, nullptr, nullptr,
                                             l1g + l * kD, l1b + l * kD, xn, xbf);
    { float* t = xc; xc = xn; xn = t; }

    gemm_bf16_wmma<2><<<gFF, blk, 0, stream>>>(xbf, W1T + l * Wdf, b1 + l * kFF, h1, kM, kFF, kD);
    gemm_bf16_wmma<0><<<gD, blk, 0, stream>>>(h1, W2T + l * Wdf, b2 + l * kD, t0, kM, kD, kFF);
    layernorm_kernel<<<kM, blk, 0, stream>>>(t0, xc, nullptr, nullptr,
                                             l2g + l * kD, l2b + l * kD, xn, xbf);
    { float* t = xc; xc = xn; xn = t; }
  }

  classifier_kernel<<<dim3(kB), blk, 0, stream>>>(xc, Wc, bc, (float*)d_out);
}
